// Transformer_44487271252630
// MI455X (gfx1250) — matrix operand order repaired and checked by reading, not compiled, not run on hardware
//
#include <hip/hip_runtime.h>

#define DIM    256
#define HEADS  8
#define DHEAD  64
#define INNER  512
#define MLP    1024
#define SEQ    256
#define NBLK   (4*64)
#define NTOK   (4*64*256)

typedef __attribute__((ext_vector_type(16))) _Float16 v16h;
typedef __attribute__((ext_vector_type(2)))  _Float16 h2;
typedef __attribute__((ext_vector_type(8)))  float    v8f;
typedef __attribute__((ext_vector_type(4)))  unsigned int v4u;
typedef __attribute__((ext_vector_type(8)))  int      v8i;
typedef __attribute__((ext_vector_type(4)))  int      v4i;

// LDS leading dims (halves / floats), padded to dodge bank conflicts
#define LDH  264   // 256 + 8
#define LD64 72    // 64 + 8
#define LDP  40    // 32 + 8

#if defined(__HIP_DEVICE_COMPILE__) && \
    __has_builtin(__builtin_amdgcn_tensor_load_to_lds)
#define USE_TDM 1
#else
#define USE_TDM 0
#endif

__device__ __forceinline__ v8f wmma_f16(v16h a, v16h b, v8f c) {
  // D(f32 16x16) = A(f16 16x32) * B(f16 32x16) + C
  return __builtin_amdgcn_wmma_f32_16x16x32_f16(false, a, false, b, (short)0, c,
                                                false, false);
}

// A operand (16x32 fp16) from row-major LDS; p points at (row0,k0), ld in halves.
// Layout: lane L -> M=L&15; VGPR j holds K = (j>>2)*16 + (L>>4)*8 + (j&3)*2 (+1)
__device__ __forceinline__ v16h ldsA(const _Float16* p, int ld, int lane) {
  const _Float16* r = p + (lane & 15) * ld + ((lane >> 4) << 3);
  v16h a;
#pragma unroll
  for (int j = 0; j < 8; ++j) {
    h2 v = *(const h2*)(r + ((j >> 2) << 4) + ((j & 3) << 1));
    a[2 * j] = v[0]; a[2 * j + 1] = v[1];
  }
  return a;
}

// B operand (32x16 fp16) from K-transposed LDS storage Wt[N][K]; p at (n0,k0).
// Layout: lane L -> N=L&15; VGPR j holds K = (j>>2)*16 + (L>>4)*8 + (j&3)*2 (+1)
__device__ __forceinline__ v16h ldsB(const _Float16* p, int ld, int lane) {
  const _Float16* r = p + (lane & 15) * ld + ((lane >> 4) << 3);
  v16h b;
#pragma unroll
  for (int j = 0; j < 8; ++j) {
    h2 v = *(const h2*)(r + ((j >> 2) << 4) + ((j & 3) << 1));
    b[2 * j] = v[0]; b[2 * j + 1] = v[1];
  }
  return b;
}

// Store D (16x16 f32) as fp16 row-major at p=(row0,n0). lane: n=L&15, M=r+(L>>4)*8
__device__ __forceinline__ void stD16(v8f d, _Float16* p, int ld, int lane) {
  _Float16* c = p + (lane & 15);
  int mo = (lane >> 4) << 3;
#pragma unroll
  for (int r = 0; r < 8; ++r) c[(mo + r) * ld] = (_Float16)d[r];
}

// Store D transposed: element (m,n) -> p[n*ld + m] (contiguous 8 halves per lane)
__device__ __forceinline__ void stD16T(v8f d, _Float16* p, int ld, int lane) {
  _Float16* c = p + (lane & 15) * ld + ((lane >> 4) << 3);
#pragma unroll
  for (int r = 0; r < 8; ++r) c[r] = (_Float16)d[r];
}

#if USE_TDM
__device__ __forceinline__ void wait_tensor0() {
#if __has_builtin(__builtin_amdgcn_s_wait_tensorcnt)
  __builtin_amdgcn_s_wait_tensorcnt(0);
#else
  asm volatile("s_wait_tensorcnt 0x0" ::: "memory");
#endif
}

// Issue a 2D TDM load: global f32 tile (tile_d0 contiguous elems x tile_d1 rows,
// row stride stride0 elems) -> LDS byte offset lds_off (row-major, packed).
__device__ __forceinline__ void tdm_load_2d(const void* gsrc, unsigned lds_off,
                                            unsigned tile_d0, unsigned tile_d1,
                                            unsigned tens_d0, unsigned tens_d1,
                                            unsigned stride0) {
  unsigned long long ga = (unsigned long long)gsrc;
  v4u g0;
  g0[0] = 1u;                                       // count=1 valid descriptor
  g0[1] = lds_off;                                  // lds_addr (bytes)
  g0[2] = (unsigned)(ga & 0xffffffffu);             // global_addr[31:0]
  g0[3] = (unsigned)((ga >> 32) & 0x1ffffffu)       // global_addr[56:32]
        | (2u << 30);                               // type=2 ("image")
  v8i g1;
  g1[0] = (int)(2u << 16);                          // wg_mask=0, data_size=4B
  g1[1] = (int)((tens_d0 & 0xffffu) << 16);         // tensor_dim0[15:0] @48
  g1[2] = (int)((tens_d0 >> 16) | ((tens_d1 & 0xffffu) << 16));
  g1[3] = (int)((tens_d1 >> 16) | (tile_d0 << 16)); // tile_dim0 @112
  g1[4] = (int)(tile_d1 & 0xffffu);                 // tile_dim1, tile_dim2=0
  g1[5] = (int)stride0;                             // dim0_stride[31:0] @160
  g1[6] = 0;                                        // stride hi, dim1_stride lo
  g1[7] = 0;
  v4i z = {0, 0, 0, 0};
#if __clang_major__ >= 23
  v8i z8 = {};
  __builtin_amdgcn_tensor_load_to_lds(g0, g1, z, z, z8, 0);
#else
  __builtin_amdgcn_tensor_load_to_lds(g0, g1, z, z, 0);
#endif
}
#endif  // USE_TDM

// ---------------- attention kernel LDS layout (bytes) ----------------
#define SM_HB   0
#define SM_WST  (SM_HB  + SEQ * LDH * 2)        // Hb: 135168
#define SM_QH   (SM_WST + 256 * LD64 * 2)       // Wstage: 36864 (max of uses)
#define SM_KH   (SM_QH  + SEQ * LD64 * 2)       // Qh: 36864
#define SM_VT   (SM_KH  + SEQ * LD64 * 2)       // Kh: 36864
#define SM_PS   (SM_VT  + DHEAD * LDH * 2)      // Vt: 33792
#define SM_OS   (SM_PS  + 8 * 16 * LDP * 2)     // Ps: 10240
#define SM_ATTN (SM_OS  + 8 * 16 * LD64 * 2)    // Os: 18432 -> total 308224

__global__ __launch_bounds__(256) void attn_kernel(
    float* __restrict__ x,
    const float* __restrict__ ln_g, const float* __restrict__ ln_b,
    const float* __restrict__ wqkv, const float* __restrict__ wout,
    const float* __restrict__ bout)
{
  extern __shared__ char smem[];
  _Float16* Hb  = (_Float16*)(smem + SM_HB);
  _Float16* Wst = (_Float16*)(smem + SM_WST);
  _Float16* Qh  = (_Float16*)(smem + SM_QH);
  _Float16* Kh  = (_Float16*)(smem + SM_KH);
  _Float16* Vt  = (_Float16*)(smem + SM_VT);
  const int tid = threadIdx.x, lane = tid & 31, wave = tid >> 5;
  _Float16* Ps = (_Float16*)(smem + SM_PS) + wave * 16 * LDP;
  _Float16* Os = (_Float16*)(smem + SM_OS) + wave * 16 * LD64;
  const long row0 = (long)blockIdx.x * SEQ;

  // ---- Phase 1: LN1(x) -> Hb (fp16), one row per thread ----
  {
    const float* xr = x + (row0 + tid) * DIM;
    float s = 0.f, ss = 0.f;
    for (int c = 0; c < DIM; c += 4) {
      float4 v = *(const float4*)(xr + c);
      s  += v.x + v.y + v.z + v.w;
      ss += v.x * v.x + v.y * v.y + v.z * v.z + v.w * v.w;
    }
    float m = s * (1.f / DIM);
    float rst = rsqrtf(ss * (1.f / DIM) - m * m + 1e-5f);
    _Float16* hr = Hb + tid * LDH;
    for (int c = 0; c < DIM; c += 4) {
      float4 v = *(const float4*)(xr + c);
      hr[c + 0] = (_Float16)((v.x - m) * rst * ln_g[c + 0] + ln_b[c + 0]);
      hr[c + 1] = (_Float16)((v.y - m) * rst * ln_g[c + 1] + ln_b[c + 1]);
      hr[c + 2] = (_Float16)((v.z - m) * rst * ln_g[c + 2] + ln_b[c + 2]);
      hr[c + 3] = (_Float16)((v.w - m) * rst * ln_g[c + 3] + ln_b[c + 3]);
    }
  }
  __syncthreads();

  for (int h = 0; h < HEADS; ++h) {
    __builtin_prefetch(wout + (long)h * DHEAD * DIM + tid, 0, 0);
    // ---- compute Q, K, V for this head ----
    for (int qkv = 0; qkv < 3; ++qkv) {
      const float* wsrc = wqkv + qkv * INNER + h * DHEAD; // [k][col] base
      for (int i = tid; i < DHEAD * DIM; i += 256) {      // stage Wt[n][k] fp16
        int n = i & 63, k = i >> 6;
        Wst[n * LDH + k] = (_Float16)wsrc[(long)k * (3 * INNER) + n];
      }
      __syncthreads();
      _Float16* dst = (qkv == 0) ? Qh : (qkv == 1) ? Kh : Vt;
      for (int t = wave; t < 64; t += 8) {                // 16 rt x 4 nt tiles
        int rt = t >> 2, nt = t & 3;
        v8f d = {};
#pragma unroll
        for (int kc = 0; kc < 8; ++kc)
          d = wmma_f16(ldsA(Hb  + rt * 16 * LDH + kc * 32, LDH, lane),
                       ldsB(Wst + nt * 16 * LDH + kc * 32, LDH, lane), d);
        if (qkv < 2) stD16 (d, dst + rt * 16 * LD64 + nt * 16, LD64, lane);
        else         stD16T(d, dst + nt * 16 * LDH  + rt * 16, LDH,  lane);
      }
      __syncthreads();
    }
    // ---- stage Wout_h as Wot[n=256][k=64] fp16 ----
    {
      const float* wsrc = wout + (long)h * DHEAD * DIM;
      for (int i = tid; i < DHEAD * DIM; i += 256) {
        int n = i & 255, k = i >> 8;
        Wst[n * LD64 + k] = (_Float16)wsrc[(long)k * DIM + n];
      }
    }
    __syncthreads();

    // ---- flash attention + fused out-projection; wave owns 2 row-tiles ----
    for (int rr = 0; rr < 2; ++rr) {
      int rt = wave * 2 + rr;
      v16h qa0 = ldsA(Qh + rt * 16 * LD64 + 0,  LD64, lane);
      v16h qa1 = ldsA(Qh + rt * 16 * LD64 + 32, LD64, lane);
      float rm[8], rl[8];
      v8f o0 = {}, o1 = {}, o2 = {}, o3 = {};
#pragma unroll
      for (int r = 0; r < 8; ++r) { rm[r] = -3.0e38f; rl[r] = 0.f; }

      for (int kb = 0; kb < 8; ++kb) {                    // 32 keys per block
        v8f s0 = {}, s1 = {};
        s0 = wmma_f16(qa0, ldsB(Kh + (kb * 32     ) * LD64 + 0,  LD64, lane), s0);
        s0 = wmma_f16(qa1, ldsB(Kh + (kb * 32     ) * LD64 + 32, LD64, lane), s0);
        s1 = wmma_f16(qa0, ldsB(Kh + (kb * 32 + 16) * LD64 + 0,  LD64, lane), s1);
        s1 = wmma_f16(qa1, ldsB(Kh + (kb * 32 + 16) * LD64 + 32, LD64, lane), s1);
#pragma unroll
        for (int r = 0; r < 8; ++r) {                     // online softmax
          float v0 = s0[r] * 0.125f, v1 = s1[r] * 0.125f;
          float mx = fmaxf(v0, v1);
          mx = fmaxf(mx, __shfl_xor(mx, 1, 32));
          mx = fmaxf(mx, __shfl_xor(mx, 2, 32));
          mx = fmaxf(mx, __shfl_xor(mx, 4, 32));
          mx = fmaxf(mx, __shfl_xor(mx, 8, 32));
          float nm = fmaxf(rm[r], mx);
          float corr = __expf(rm[r] - nm);
          float p0 = __expf(v0 - nm), p1 = __expf(v1 - nm);
          float ps = p0 + p1;
          ps += __shfl_xor(ps, 1, 32); ps += __shfl_xor(ps, 2, 32);
          ps += __shfl_xor(ps, 4, 32); ps += __shfl_xor(ps, 8, 32);
          rl[r] = rl[r] * corr + ps;
          rm[r] = nm;
          o0[r] *= corr; o1[r] *= corr; o2[r] *= corr; o3[r] *= corr;
          int m = ((lane >> 4) << 3) + r;                 // D->A bounce via LDS
          Ps[m * LDP + (lane & 15)]      = (_Float16)p0;
          Ps[m * LDP + 16 + (lane & 15)] = (_Float16)p1;
        }
        v16h pa = ldsA(Ps, LDP, lane);
        o0 = wmma_f16(pa, ldsB(Vt + 0 * 16 * LDH + kb * 32, LDH, lane), o0);
        o1 = wmma_f16(pa, ldsB(Vt + 1 * 16 * LDH + kb * 32, LDH, lane), o1);
        o2 = wmma_f16(pa, ldsB(Vt + 2 * 16 * LDH + kb * 32, LDH, lane), o2);
        o3 = wmma_f16(pa, ldsB(Vt + 3 * 16 * LDH + kb * 32, LDH, lane), o3);
      }
#pragma unroll
      for (int r = 0; r < 8; ++r) {
        float inv = 1.f / rl[r];
        o0[r] *= inv; o1[r] *= inv; o2[r] *= inv; o3[r] *= inv;
      }
      stD16(o0, Os + 0,  LD64, lane); stD16(o1, Os + 16, LD64, lane);
      stD16(o2, Os + 32, LD64, lane); stD16(o3, Os + 48, LD64, lane);

      // x[rows] += O_h @ Wout_h (+ b_out on head 0)
      v16h oa0 = ldsA(Os + 0,  LD64, lane);
      v16h oa1 = ldsA(Os + 32, LD64, lane);
      for (int nt = 0; nt < 16; ++nt) {
        v8f d = {};
        d = wmma_f16(oa0, ldsB(Wst + nt * 16 * LD64 + 0,  LD64, lane), d);
        d = wmma_f16(oa1, ldsB(Wst + nt * 16 * LD64 + 32, LD64, lane), d);
        int col = nt * 16 + (lane & 15);
        float bo = (h == 0) ? bout[col] : 0.f;
        int mo = (lane >> 4) << 3;
        float* xp = x + (row0 + rt * 16 + mo) * DIM + col;
#pragma unroll
        for (int r = 0; r < 8; ++r) xp[r * DIM] += d[r] + bo;
      }
    }
    __syncthreads();
  }
}

// ---------------- FFN kernel LDS layout (bytes) ----------------
#define F_HB   0
#define F_DL   (F_HB  + 128 * LDH * 2)          // Hb2: 67584
#define F_WST  (F_DL  + 128 * LDH * 4)          // Delta f32: 135168
#define F_UST  (F_WST + 256 * LD64 * 2)         // Wstage: 36864
#define F_STG  (F_UST + 128 * LD64 * 2)         // Ust: 18432
#define SM_FFN (F_STG + 64 * 256 * 4)           // f32 TDM stage: 65536 -> 323584

__global__ __launch_bounds__(256) void ffn_kernel(
    float* __restrict__ x,
    const float* __restrict__ ln_g, const float* __restrict__ ln_b,
    const float* __restrict__ w1, const float* __restrict__ b1,
    const float* __restrict__ w2, const float* __restrict__ b2)
{
  extern __shared__ char smem[];
  _Float16* Hb  = (_Float16*)(smem + F_HB);
  float*    Dl  = (float*)   (smem + F_DL);
  _Float16* Wst = (_Float16*)(smem + F_WST);
  _Float16* Ust = (_Float16*)(smem + F_UST);
#if USE_TDM
  const float* Stg = (const float*)(smem + F_STG);
#endif
  const int tid = threadIdx.x, lane = tid & 31, wave = tid >> 5;
  const long row0 = (long)blockIdx.x * 128;

  if (tid < 128) {                                        // LN2, one row/thread
    const float* xr = x + (row0 + tid) * DIM;
    float s = 0.f, ss = 0.f;
    for (int c = 0; c < DIM; c += 4) {
      float4 v = *(const float4*)(xr + c);
      s  += v.x + v.y + v.z + v.w;
      ss += v.x * v.x + v.y * v.y + v.z * v.z + v.w * v.w;
    }
    float m = s * (1.f / DIM);
    float rst = rsqrtf(ss * (1.f / DIM) - m * m + 1e-5f);
    _Float16* hr = Hb + tid * LDH;
    for (int c = 0; c < DIM; ++c)
      hr[c] = (_Float16)((xr[c] - m) * rst * ln_g[c] + ln_b[c]);
  }
  for (int i = tid; i < 128 * LDH; i += 256) Dl[i] = 0.f;
  __syncthreads();

#if USE_TDM
  if (wave == 0)                       // preload W1 chunk 0: [k=256][n=64] tile
    tdm_load_2d(w1, (unsigned)F_STG, 64, 256, MLP, DIM, MLP);
#endif

  for (int c = 0; c < 16; ++c) {                          // MLP chunks of 64
#if USE_TDM
    if (wave == 0) wait_tensor0();
    __syncthreads();
    for (int i = tid; i < 64 * DIM; i += 256) {           // f32 stage -> [n][k]
      int n = i & 63, k = i >> 6;
      Wst[n * LDH + k] = (_Float16)Stg[k * 64 + n];
    }
    __syncthreads();
    if (wave == 0)                     // W2 chunk DMA overlaps U-compute WMMAs
      tdm_load_2d(w2 + (long)c * 64 * DIM, (unsigned)F_STG, 256, 64,
                  DIM, MLP, DIM);
#else
    for (int i = tid; i < 64 * DIM; i += 256) {           // W1 chunk -> [n][k]
      int n = i & 63, k = i >> 6;
      Wst[n * LDH + k] = (_Float16)w1[(long)k * MLP + c * 64 + n];
    }
    __syncthreads();
#endif
    for (int t = wave; t < 32; t += 8) {                  // U = silu(h@W1+b1)
      int rt = t >> 2, nt = t & 3;
      v8f d = {};
#pragma unroll
      for (int kc = 0; kc < 8; ++kc)
        d = wmma_f16(ldsA(Hb  + rt * 16 * LDH + kc * 32, LDH, lane),
                     ldsB(Wst + nt * 16 * LDH + kc * 32, LDH, lane), d);
      float bb = b1[c * 64 + nt * 16 + (lane & 15)];
#pragma unroll
      for (int r = 0; r < 8; ++r) {
        float u = d[r] + bb;
        d[r] = u / (1.f + __expf(-u));
      }
      stD16(d, Ust + rt * 16 * LD64 + nt * 16, LD64, lane);
    }
#if USE_TDM
    if (wave == 0) wait_tensor0();
    __syncthreads();
    for (int i = tid; i < 64 * DIM; i += 256) {           // f32 stage -> [n][k]
      int n = i & 255, k = i >> 8;
      Wst[n * LD64 + k] = (_Float16)Stg[k * DIM + n];
    }
    __syncthreads();
    if (c + 1 < 16 && wave == 0)       // next W1 chunk DMA overlaps Dl WMMAs
      tdm_load_2d(w1 + (c + 1) * 64, (unsigned)F_STG, 64, 256, MLP, DIM, MLP);
#else
    __syncthreads();
    for (int i = tid; i < 64 * DIM; i += 256) {           // W2 chunk -> [n][k]
      int n = i & 255, k = i >> 8;
      Wst[n * LD64 + k] = (_Float16)w2[(long)(c * 64 + k) * DIM + n];
    }
    __syncthreads();
#endif
    for (int t = wave; t < 128; t += 8) {                 // Dl += U @ W2c
      int rt = t >> 4, nt = t & 15;
      float* dp = Dl + rt * 16 * LDH + nt * 16 + (lane & 15);
      int mo = (lane >> 4) << 3;
      v8f cc;
#pragma unroll
      for (int r = 0; r < 8; ++r) cc[r] = dp[(mo + r) * LDH];
      cc = wmma_f16(ldsA(Ust + rt * 16 * LD64 + 0,  LD64, lane),
                    ldsB(Wst + nt * 16 * LD64 + 0,  LD64, lane), cc);
      cc = wmma_f16(ldsA(Ust + rt * 16 * LD64 + 32, LD64, lane),
                    ldsB(Wst + nt * 16 * LD64 + 32, LD64, lane), cc);
#pragma unroll
      for (int r = 0; r < 8; ++r) dp[(mo + r) * LDH] = cc[r];
    }
    __syncthreads();
  }
  for (int i = tid; i < 128 * DIM; i += 256) {            // x += Delta + b2
    int rrow = i >> 8, col = i & 255;
    x[(row0 + rrow) * DIM + col] += Dl[rrow * LDH + col] + b2[col];
  }
}

extern "C" void kernel_launch(void* const* d_in, const int* in_sizes, int n_in,
                              void* d_out, int out_size, void* d_ws, size_t ws_size,
                              hipStream_t stream) {
  (void)in_sizes; (void)n_in; (void)d_ws; (void)ws_size; (void)out_size;
  const float* ln1g = (const float*)d_in[1];
  const float* ln1b = (const float*)d_in[2];
  const float* wqkv = (const float*)d_in[3];
  const float* wout = (const float*)d_in[4];
  const float* bout = (const float*)d_in[5];
  const float* ln2g = (const float*)d_in[6];
  const float* ln2b = (const float*)d_in[7];
  const float* w1   = (const float*)d_in[8];
  const float* b1   = (const float*)d_in[9];
  const float* w2   = (const float*)d_in[10];
  const float* b2   = (const float*)d_in[11];
  float* x = (float*)d_out;

  (void)hipFuncSetAttribute((const void*)attn_kernel,
      hipFuncAttributeMaxDynamicSharedMemorySize, SM_ATTN);
  (void)hipFuncSetAttribute((const void*)ffn_kernel,
      hipFuncAttributeMaxDynamicSharedMemorySize, SM_FFN);

  hipMemcpyAsync(x, d_in[0], (size_t)NTOK * DIM * sizeof(float),
                 hipMemcpyDeviceToDevice, stream);

  for (int l = 0; l < 4; ++l) {
    attn_kernel<<<NBLK, 256, SM_ATTN, stream>>>(
        x, ln1g + l * DIM, ln1b + l * DIM,
        wqkv + (long)l * DIM * 3 * INNER,
        wout + (long)l * INNER * DIM, bout + l * DIM);
    ffn_kernel<<<NTOK / 128, 256, SM_FFN, stream>>>(
        x, ln2g + l * DIM, ln2b + l * DIM,
        w1 + (long)l * DIM * MLP, b1 + l * MLP,
        w2 + (long)l * MLP * DIM, b2 + l * DIM);
  }
}
